// DimeNetPlus_36352603194133
// MI455X (gfx1250) — matrix operand order repaired and checked by reading, not compiled, not run on hardware
//
#include <hip/hip_runtime.h>

// ---------------------------------------------------------------------------
// DimeNet++ forward for MI455X (gfx1250, wave32, WMMA).
// - All dense 128-wide GEMMs: v_wmma_f32_16x16x32_f16 (f16 in, f32 acc).
// - Weights staged in LDS (32KB static GEMM / 256KB dynamic einsum); B
//   fragments come from ds_load_b128 with immediate offsets off 8
//   precomputed per-n-tile base pointers, software-pipelined 2-deep so
//   loads overlap WMMAs (no dscnt bubbles / WAR hazard nops).
// - Empty asm memory fences bound the scheduler's load-hoisting scope so
//   B fragments never spill to scratch; grid-stride loops amortize fills.
// - GEMM fusion flags are template parameters; silu uses v_rcp_f32.
// - Per-row sbf scaling via v_pk_mul_f16 (2-wide half vectors).
// - Gathers fused into GEMM A-loads via row_map (+ global_prefetch of the
//   next tile); scatter-adds fused via global_atomic_add_f32 from registers.
// ---------------------------------------------------------------------------

typedef __attribute__((ext_vector_type(16))) _Float16 v16h;
typedef __attribute__((ext_vector_type(8)))  _Float16 v8h;
typedef __attribute__((ext_vector_type(2)))  _Float16 h2;
typedef __attribute__((ext_vector_type(8)))  float    v8f;

#define H 128
#define NRAD 6
#define NSBF 42
#define NBIL 8
#define NBLK 6

#define SCHED_FENCE() asm volatile("" ::: "memory")

__device__ __forceinline__ float silu_f(float v) {
    return v * __builtin_amdgcn_rcpf(1.0f + __expf(-v));   // v_rcp_f32, no div
}

__device__ __forceinline__ v16h ldB(const _Float16* p) {   // 2x b128 load
    v8h b0 = *(const v8h*)p;
    v8h b1 = *(const v8h*)(p + 8);
    v16h bb;
#pragma unroll
    for (int e = 0; e < 8; ++e) { bb[e] = b0[e]; bb[8 + e] = b1[e]; }
    return bb;
}

__device__ __forceinline__ v16h ldA(const _Float16* p) {   // K-halves 0..15
    v8h lo = *(const v8h*)p;
    v8h hi = *(const v8h*)(p + 8);
    v16h a;
#pragma unroll
    for (int e = 0; e < 8; ++e) { a[e] = lo[e]; a[8 + e] = hi[e]; }
    return a;
}

// per-lane scalar scale of a 16-half fragment -> 8x v_pk_mul_f16
__device__ __forceinline__ v16h scale16(v16h a, _Float16 s) {
    h2 sp; sp[0] = s; sp[1] = s;
    union { v16h v; h2 p[8]; } ua, ur;
    ua.v = a;
#pragma unroll
    for (int i = 0; i < 8; ++i) ur.p[i] = ua.p[i] * sp;
    return ur.v;
}

// ---------------------------------------------------------------------------
// 128x128 WMMA GEMM (K = 128), fusion flags as template params:
//   acc = [bias[n]] + [pre[row,n]] + sum_k A[src_row, k] * Wt[n, k]
//   v = [silu](acc); [v *= mul];  [v += post];  [Cf]=v; [Ch]=(f16)v
// Wt row n at Wt + n*wstride (pre-transposed f16); staged into 32KB LDS.
// One wave per 16x128 row tile, grid-stride over tiles. blockDim = 256.
// ---------------------------------------------------------------------------
template<bool MAP, bool BIAS, bool PRE, bool ACT, bool MUL, bool POST,
         bool WF, bool WH>
__global__ __launch_bounds__(256)
void k_gemm128(const _Float16* __restrict__ A,
               const int* __restrict__ row_map,
               const _Float16* __restrict__ Wt, int wstride,
               const float* __restrict__ bias,
               const float* __restrict__ pre,
               const float* __restrict__ mul_src,
               const float* __restrict__ post,
               float* __restrict__ Cf,
               _Float16* __restrict__ Ch,
               int M)
{
    __shared__ _Float16 sW[H * H];              // 32 KB
    {
        v8h* d = (v8h*)sW;                      // 2048 v8h
        for (int idx = threadIdx.x; idx < (H * H) / 8; idx += 256) {
            int n  = idx >> 4;                  // 16 v8h per 128-wide row
            int ko = (idx & 15) * 8;
            d[idx] = *(const v8h*)(Wt + (long)n * wstride + ko);
        }
    }
    __syncthreads();

    const int wave = threadIdx.x >> 5;
    const int lane = threadIdx.x & 31;
    const int half = lane >> 4;                 // 0 or 1
    const int lm   = lane & 15;
    const int tiles = M >> 4;                   // M % 16 == 0 for all callers

    for (int tl = blockIdx.x * 8 + wave; tl < tiles; tl += gridDim.x * 8) {
        SCHED_FENCE();                          // no cross-tile load hoisting
        const long row0 = (long)tl * 16;
        const long arow = row0 + lm;
        const long asrc = MAP ? (long)row_map[arow] : arow;
        const _Float16* Abase = A + asrc * H + half * 16;

        // all 4 K-chunk A fragments in registers (reused by 8 n-tiles)
        v16h afr[4];
#pragma unroll
        for (int kc = 0; kc < 4; ++kc) afr[kc] = ldA(Abase + kc * 32);

        // prefetch next grid-stride tile's A row (covers gather latency)
        {
            const int tln = tl + gridDim.x * 8;
            if (tln < tiles) {
                const long arown = (long)tln * 16 + lm;
                const long asrcn = MAP ? (long)row_map[arown] : arown;
                __builtin_prefetch(A + asrcn * H, 0, 3);
            }
        }

        for (int nt = 0; nt < 8; ++nt) {
            SCHED_FENCE();                      // <=4 B fragments in flight
            const int n = nt * 16 + lm;
            v8f acc;
            const float binit = BIAS ? bias[n] : 0.0f;
#pragma unroll
            for (int r = 0; r < 8; ++r) acc[r] = binit;
            if (PRE) {
#pragma unroll
                for (int r = 0; r < 8; ++r)
                    acc[r] += pre[(row0 + half * 8 + r) * H + n];
            }
            const _Float16* Brow = sW + n * H + half * 16;   // LDS
            v16h bfr[4];
#pragma unroll
            for (int kc = 0; kc < 4; ++kc) bfr[kc] = ldB(Brow + kc * 32);
#pragma unroll
            for (int kc = 0; kc < 4; ++kc)
                acc = __builtin_amdgcn_wmma_f32_16x16x32_f16(
                    false, afr[kc], false, bfr[kc], (short)0, acc, false, false);
#pragma unroll
            for (int r = 0; r < 8; ++r) {
                const long row = row0 + half * 8 + r;
                float v = acc[r];
                if (ACT)  v = silu_f(v);
                if (MUL)  v *= mul_src[row * H + n];
                if (POST) v += post[row * H + n];
                if (WF) Cf[row * H + n] = v;
                if (WH) Ch[row * H + n] = (_Float16)v;
            }
        }
    }
}

// ---------------------------------------------------------------------------
// Bilinear einsum + scatter:
//   m[t,i] = sum_j sbf_p[t,j] * sum_l x_kj[idx_kj[t], l] * W[i, j, l]
//   out[idx_ji[t], i] += m[t,i]     (atomic f32)
// Full W16 (f16 [128][NBIL*128], 256KB) staged in dynamic LDS per workgroup,
// grid-stride tile loop amortizes the fill. blockDim = 512 (16 waves).
// B addressed via 8 precomputed tile-invariant n-tile base pointers
// (in-loop DS offsets fit the 16-bit immediate) and software-pipelined
// 2-deep so ds_loads overlap WMMAs.
// ---------------------------------------------------------------------------
__global__ __launch_bounds__(512)
void k_einsum_scatter(const _Float16* __restrict__ Xkj,
                      const int* __restrict__ idx_kj,
                      const int* __restrict__ idx_ji,
                      const float* __restrict__ sbf_p,
                      const _Float16* __restrict__ W16,
                      float* __restrict__ out,
                      int T)
{
    extern __shared__ _Float16 sW[];            // H * NBIL * H = 256 KB
    {
        v8h*       d = (v8h*)sW;                // 16384 v8h
        const v8h* s = (const v8h*)W16;         // layout preserved: contiguous
        for (int idx = threadIdx.x; idx < (H * NBIL * H) / 8; idx += 512)
            d[idx] = s[idx];
    }
    __syncthreads();

    const int wave = threadIdx.x >> 5;
    const int lane = threadIdx.x & 31;
    const int half = lane >> 4;
    const int lm   = lane & 15;
    const int tiles = T >> 4;                   // T % 16 == 0

    // tile-invariant per-n-tile LDS base pointers (big n*2048B component
    // hoisted once; in-loop offsets j*256 + kc*64 + 16 fit DS immediates)
    const _Float16* ntbase[8];
#pragma unroll
    for (int nt = 0; nt < 8; ++nt)
        ntbase[nt] = sW + (long)(nt * 16 + lm) * (NBIL * H) + half * 16;

    for (int tl = blockIdx.x * 16 + wave; tl < tiles; tl += gridDim.x * 16) {
        SCHED_FENCE();                          // no cross-tile load hoisting
        const long t0   = (long)tl * 16;
        const long trow = t0 + lm;

        const long asrc = idx_kj[trow];
        const _Float16* Abase = Xkj + asrc * H + half * 16;

        v16h afr[4];
#pragma unroll
        for (int kc = 0; kc < 4; ++kc) afr[kc] = ldA(Abase + kc * 32);
        v8f s = *(const v8f*)(sbf_p + trow * NBIL);

        v8f acc[8];
#pragma unroll
        for (int nt = 0; nt < 8; ++nt)
#pragma unroll
            for (int r = 0; r < 8; ++r) acc[nt][r] = 0.0f;

#pragma unroll
        for (int j = 0; j < NBIL; ++j) {
            const _Float16 sh = (_Float16)s[j];
#pragma unroll
            for (int kc = 0; kc < 4; ++kc) {
                SCHED_FENCE();                  // bound in-flight B fragments
                const int joff = j * H + kc * 32;
                const v16h as = scale16(afr[kc], sh);     // 8x v_pk_mul_f16
                // 2-deep software pipeline: load nt+1 before WMMA consumes nt
                v16h bcur = ldB(ntbase[0] + joff);
#pragma unroll
                for (int nt = 0; nt < 8; ++nt) {
                    v16h bnext = bcur;
                    if (nt < 7) bnext = ldB(ntbase[nt + 1] + joff);
                    acc[nt] = __builtin_amdgcn_wmma_f32_16x16x32_f16(
                        false, as, false, bcur, (short)0, acc[nt],
                        false, false);
                    bcur = bnext;
                }
            }
        }

        int dest[8];
#pragma unroll
        for (int r = 0; r < 8; ++r) dest[r] = idx_ji[t0 + half * 8 + r];
#pragma unroll
        for (int nt = 0; nt < 8; ++nt) {
            const int n = nt * 16 + lm;
#pragma unroll
            for (int r = 0; r < 8; ++r)
                atomicAdd(out + (long)dest[r] * H + n, acc[nt][r]);
        }
    }
}

// ---------------------------------------------------------------------------
// Small-K dense layer (K = 6 or 42): out = act(In @ W + b), optional f16 copy
// ---------------------------------------------------------------------------
__global__ void k_small(const float* __restrict__ In, int Ks,
                        const float* __restrict__ W, int Nn,
                        const float* __restrict__ bias, int act,
                        float* __restrict__ Of, _Float16* __restrict__ Oh,
                        long MN)
{
    long t = (long)blockIdx.x * blockDim.x + threadIdx.x;
    if (t >= MN) return;
    long m = t / Nn;
    int  n = (int)(t - m * Nn);
    float a = bias ? bias[n] : 0.0f;
    for (int k = 0; k < Ks; ++k) a += In[m * Ks + k] * W[(long)k * Nn + n];
    if (act) a = silu_f(a);
    if (Of) Of[t] = a;
    if (Oh) Oh[t] = (_Float16)a;
}

// Bessel RBF with smooth envelope -> rbf [E, 6]
__global__ void k_rbf(const float* __restrict__ dist,
                      const float* __restrict__ freq,
                      float* __restrict__ rbf, int E)
{
    int e = blockIdx.x * blockDim.x + threadIdx.x;
    if (e >= E) return;
    float d = dist[e] * 0.2f;               // dist / 5.0
    float d2 = d * d, d4 = d2 * d2, d5 = d4 * d;
    float env = 1.0f / d + (-28.0f) * d5 + 48.0f * d5 * d + (-21.0f) * d5 * d2;
#pragma unroll
    for (int r = 0; r < NRAD; ++r)
        rbf[(long)e * NRAD + r] = env * __sinf(freq[r] * d);
}

// node embedding concat -> f16 [N, 128]
__global__ void k_node(const int* __restrict__ x,
                       const float* __restrict__ resi,
                       const float* __restrict__ atom,
                       _Float16* __restrict__ nodeh, long total)
{
    long t = (long)blockIdx.x * blockDim.x + threadIdx.x;
    if (t >= total) return;
    int v = (int)(t >> 7), c = (int)(t & 127);
    float val = (c < 64) ? resi[(long)x[2 * v] * 64 + c]
                         : atom[(long)x[2 * v + 1] * 64 + (c - 64)];
    nodeh[t] = (_Float16)val;
}

__global__ void k_cvt(const float* __restrict__ s, _Float16* __restrict__ d,
                      long n)
{
    long t = (long)blockIdx.x * blockDim.x + threadIdx.x;
    if (t < n) d[t] = (_Float16)s[t];
}

// transpose + f16 convert: W [K][Nn] f32 -> Wt [Nn][K] f16
__global__ void k_transpose(const float* __restrict__ W,
                            _Float16* __restrict__ Wt, int K, int Nn)
{
    long t = (long)blockIdx.x * blockDim.x + threadIdx.x;
    if (t >= (long)K * Nn) return;
    long n = t / K;
    long k = t - n * K;
    Wt[t] = (_Float16)W[k * Nn + n];
}

// out-block edge->node accumulate: nacc[i[e],n] += (rbf[e] @ orw)[n] * xe[e,n]
__global__ void k_out_accum(const float* __restrict__ rbf,
                            const float* __restrict__ orw,
                            const float* __restrict__ xe,
                            const int* __restrict__ iidx,
                            float* __restrict__ nacc, long total)
{
    long t = (long)blockIdx.x * blockDim.x + threadIdx.x;
    if (t >= total) return;
    int e = (int)(t >> 7), n = (int)(t & 127);
    float rp = 0.0f;
#pragma unroll
    for (int r = 0; r < NRAD; ++r) rp += rbf[(long)e * NRAD + r] * orw[r * H + n];
    atomicAdd(nacc + (long)iidx[e] * H + n, rp * xe[t]);
}

// P[n] += dot(nf[n,:], w)
__global__ void k_proj(const float* __restrict__ nf,
                       const float* __restrict__ w,
                       float* __restrict__ P, int N)
{
    int n = blockIdx.x * blockDim.x + threadIdx.x;
    if (n >= N) return;
    float a = 0.0f;
    for (int h = 0; h < H; ++h) a += nf[(long)n * H + h] * w[h];
    P[n] += a;
}

// ---------------------------------------------------------------------------
// host orchestration
// ---------------------------------------------------------------------------
template<bool MAP, bool BIAS, bool PRE, bool ACT, bool MUL, bool POST,
         bool WF, bool WH>
static inline void launch_gemm(hipStream_t s, const _Float16* A,
                               const int* rmap, const _Float16* Wt, int wst,
                               const float* bias, const float* pre,
                               const float* mul, const float* post,
                               float* Cf, _Float16* Ch, int M)
{
    int tiles  = (M + 15) / 16;
    int blocks = (tiles + 7) / 8;
    if (blocks > 1024) blocks = 1024;          // grid-stride amortizes LDS fill
    k_gemm128<MAP, BIAS, PRE, ACT, MUL, POST, WF, WH>
        <<<blocks, 256, 0, s>>>(A, rmap, Wt, wst, bias, pre, mul, post,
                                Cf, Ch, M);
}

extern "C" void kernel_launch(void* const* d_in, const int* in_sizes, int n_in,
                              void* d_out, int out_size, void* d_ws,
                              size_t ws_size, hipStream_t stream)
{
    (void)n_in; (void)ws_size;
    const float* dist      = (const float*)d_in[0];
    const float* sbf       = (const float*)d_in[1];
    const float* freq      = (const float*)d_in[2];
    const float* resi      = (const float*)d_in[3];
    const float* atom      = (const float*)d_in[4];
    const float* emb_rbf_w = (const float*)d_in[5];
    const float* emb_rbf_b = (const float*)d_in[6];
    const float* emb_lin_w = (const float*)d_in[7];
    const float* emb_lin_b = (const float*)d_in[8];
    const float* int_rbf_w = (const float*)d_in[9];
    const float* int_sbf_w = (const float*)d_in[10];
    const float* int_kj_w  = (const float*)d_in[11];
    const float* int_kj_b  = (const float*)d_in[12];
    const float* int_ji_w  = (const float*)d_in[13];
    const float* int_ji_b  = (const float*)d_in[14];
    const float* int_W     = (const float*)d_in[15];
    const float* int_rb_w1 = (const float*)d_in[16];
    const float* int_rb_b1 = (const float*)d_in[17];
    const float* int_rb_w2 = (const float*)d_in[18];
    const float* int_rb_b2 = (const float*)d_in[19];
    const float* int_lin_w = (const float*)d_in[20];
    const float* int_lin_b = (const float*)d_in[21];
    const float* int_ra_w1 = (const float*)d_in[22];
    const float* int_ra_b1 = (const float*)d_in[23];
    const float* int_ra_w2 = (const float*)d_in[24];
    const float* int_ra_b2 = (const float*)d_in[25];
    const float* out_rbf_w = (const float*)d_in[26];
    const float* out_lins_w= (const float*)d_in[27];
    const float* out_lins_b= (const float*)d_in[28];
    const float* out_lin_w = (const float*)d_in[29];
    const int*   xnode     = (const int*)d_in[30];
    const int*   iidx      = (const int*)d_in[31];
    const int*   jidx      = (const int*)d_in[32];
    const int*   idx_kj    = (const int*)d_in[33];
    const int*   idx_ji    = (const int*)d_in[34];

    const int E = in_sizes[0];
    const int T = in_sizes[1] / NSBF;
    const int N = in_sizes[30] / 2;
    (void)out_size;

    // ---- workspace carve ----
    unsigned char* wsb = (unsigned char*)d_ws;
    size_t off = 0;
    auto alloc = [&](size_t bytes) -> void* {
        void* p = wsb + off;
        off += (bytes + 255) & ~(size_t)255;
        return p;
    };
    const long HT = (long)H * H;                      // 16384
    const long OFF_EMB = 0;                           // 384*128
    const long OFF_BLK = 49152;                       // per-block: 9*HT
    const long BLK_SZ  = 9 * HT;                      // 147456
    const long OFF_INTW = OFF_BLK + NBLK * BLK_SZ;    // 933888
    const long OFF_OUT  = OFF_INTW + (long)NBLK * H * NBIL * H; // 1720320
    const long W16_TOTAL = OFF_OUT + 21 * HT;         // 2064384

    _Float16* w16   = (_Float16*)alloc((size_t)W16_TOTAL * 2);
    float*    rbf   = (float*)alloc((size_t)E * NRAD * 4);
    float*    S     = (float*)alloc((size_t)T * NBIL * 4);
    float*    R     = (float*)alloc((size_t)E * H * 4);
    _Float16* nodeh = (_Float16*)alloc((size_t)N * H * 2);
    _Float16* rbfeh = (_Float16*)alloc((size_t)E * H * 2);
    float*    X     = (float*)alloc((size_t)E * H * 4);
    _Float16* Xh    = (_Float16*)alloc((size_t)E * H * 2);
    float*    H1    = (float*)alloc((size_t)E * H * 4);
    _Float16* H1h   = (_Float16*)alloc((size_t)E * H * 2);
    float*    H2    = (float*)alloc((size_t)E * H * 4);
    _Float16* H2h   = (_Float16*)alloc((size_t)E * H * 2);
    _Float16* T1h   = (_Float16*)alloc((size_t)E * H * 2);
    _Float16* G16   = (_Float16*)alloc((size_t)E * H * 2);
    float*    nacc  = (float*)alloc((size_t)N * H * 4);
    _Float16* M1h   = (_Float16*)alloc((size_t)N * H * 2);
    _Float16* M2h   = (_Float16*)alloc((size_t)N * H * 2);
    float*    NF    = (float*)alloc((size_t)N * H * 4);

    auto transpose = [&](const float* Wsrc, _Float16* Wdst, int K, int Nn) {
        long n = (long)K * Nn;
        k_transpose<<<(int)((n + 255) / 256), 256, 0, stream>>>(Wsrc, Wdst, K, Nn);
    };

    // ---- weight prep (f16 transpose/convert) ----
    transpose(emb_lin_w, w16 + OFF_EMB, 3 * H, H);
    for (int b = 0; b < NBLK; ++b) {
        _Float16* base = w16 + OFF_BLK + (long)b * BLK_SZ;
        transpose(int_ji_w + (long)b * HT, base + 0 * HT, H, H);
        transpose(int_kj_w + (long)b * HT, base + 1 * HT, H, H);
        transpose(int_rb_w1 + (long)b * HT, base + 2 * HT, H, H);
        transpose(int_rb_w2 + (long)b * HT, base + 3 * HT, H, H);
        transpose(int_lin_w + (long)b * HT, base + 4 * HT, H, H);
        for (int r = 0; r < 2; ++r) {
            transpose(int_ra_w1 + (long)(b * 2 + r) * HT, base + (5 + r) * HT, H, H);
            transpose(int_ra_w2 + (long)(b * 2 + r) * HT, base + (7 + r) * HT, H, H);
        }
    }
    {   // int_W: plain f32->f16 convert, layout preserved [b][i][j][l]
        long n = (long)NBLK * H * NBIL * H;
        k_cvt<<<(int)((n + 255) / 256), 256, 0, stream>>>(int_W, w16 + OFF_INTW, n);
    }
    for (int ob = 0; ob <= NBLK; ++ob)
        for (int k = 0; k < 3; ++k)
            transpose(out_lins_w + (long)(ob * 3 + k) * HT,
                      w16 + OFF_OUT + (long)(ob * 3 + k) * HT, H, H);

    // ---- basis + embedding ----
    hipMemsetAsync(d_out, 0, (size_t)N * 4, stream);
    k_rbf<<<(E + 255) / 256, 256, 0, stream>>>(dist, freq, rbf, E);
    {
        long tot = (long)N * H;
        k_node<<<(int)((tot + 255) / 256), 256, 0, stream>>>(xnode, resi, atom, nodeh, tot);
    }
    k_small<<<(int)(((long)E * H + 255) / 256), 256, 0, stream>>>(
        rbf, NRAD, emb_rbf_w, H, emb_rbf_b, 1, nullptr, rbfeh, (long)E * H);

    // xe = silu([node[i] | node[j] | rbf_e] @ emb_lin_w + b): 3 accumulating passes
    launch_gemm<true, false, false, false, false, false, true, false>(
        stream, nodeh, iidx, w16 + OFF_EMB, 3 * H,
        nullptr, nullptr, nullptr, nullptr, X, nullptr, E);
    launch_gemm<true, false, true, false, false, false, true, false>(
        stream, nodeh, jidx, w16 + OFF_EMB + H, 3 * H,
        nullptr, X, nullptr, nullptr, X, nullptr, E);
    launch_gemm<false, true, true, true, false, false, true, true>(
        stream, rbfeh, nullptr, w16 + OFF_EMB + 2 * H, 3 * H,
        emb_lin_b, X, nullptr, nullptr, X, Xh, E);

    auto out_block = [&](int ob) {
        hipMemsetAsync(nacc, 0, (size_t)N * H * 4, stream);
        long tot = (long)E * H;
        k_out_accum<<<(int)((tot + 255) / 256), 256, 0, stream>>>(
            rbf, out_rbf_w + (long)ob * NRAD * H, X, iidx, nacc, tot);
        long nt = (long)N * H;
        k_cvt<<<(int)((nt + 255) / 256), 256, 0, stream>>>(nacc, M1h, nt);
        launch_gemm<false, true, false, true, false, false, false, true>(
            stream, M1h, nullptr, w16 + OFF_OUT + (long)(ob * 3 + 0) * HT, H,
            out_lins_b + (long)(ob * 3 + 0) * H, nullptr, nullptr, nullptr,
            nullptr, M2h, N);
        launch_gemm<false, true, false, true, false, false, false, true>(
            stream, M2h, nullptr, w16 + OFF_OUT + (long)(ob * 3 + 1) * HT, H,
            out_lins_b + (long)(ob * 3 + 1) * H, nullptr, nullptr, nullptr,
            nullptr, M1h, N);
        launch_gemm<false, true, false, true, false, false, true, false>(
            stream, M1h, nullptr, w16 + OFF_OUT + (long)(ob * 3 + 2) * HT, H,
            out_lins_b + (long)(ob * 3 + 2) * H, nullptr, nullptr, nullptr,
            NF, nullptr, N);
        k_proj<<<(N + 255) / 256, 256, 0, stream>>>(NF, out_lin_w + (long)ob * H,
                                                    (float*)d_out, N);
    };

    out_block(0);

    for (int b = 0; b < NBLK; ++b) {
        _Float16* base   = w16 + OFF_BLK + (long)b * BLK_SZ;
        _Float16* intW16 = w16 + OFF_INTW + (long)b * H * NBIL * H;

        // rbf_p [E,128], sbf_p [T,8]
        k_small<<<(int)(((long)E * H + 255) / 256), 256, 0, stream>>>(
            rbf, NRAD, int_rbf_w + (long)b * NRAD * H, H, nullptr, 0, R, nullptr,
            (long)E * H);
        k_small<<<(int)(((long)T * NBIL + 255) / 256), 256, 0, stream>>>(
            sbf, NSBF, int_sbf_w + (long)b * NSBF * NBIL, NBIL, nullptr, 0, S,
            nullptr, (long)T * NBIL);

        // x_ji (into H1, f32) and x_kj = silu(..)*rbf_p (f16, for gather)
        launch_gemm<false, true, false, true, false, false, true, false>(
            stream, Xh, nullptr, base + 0 * HT, H, int_ji_b + (long)b * H,
            nullptr, nullptr, nullptr, H1, nullptr, E);
        launch_gemm<false, true, false, true, true, false, false, true>(
            stream, Xh, nullptr, base + 1 * HT, H, int_kj_b + (long)b * H,
            nullptr, R, nullptr, nullptr, G16, E);

        // bilinear einsum, scatter-add into H1 (h = x_ji + agg)
        {
            int tiles  = (T + 15) / 16;
            int blocks = (tiles + 15) / 16;
            if (blocks > 512) blocks = 512;
            k_einsum_scatter<<<blocks, 512, (size_t)H * NBIL * H * 2, stream>>>(
                G16, idx_kj, idx_ji, S, intW16, H1, T);
        }
        k_cvt<<<(int)(((long)E * H + 255) / 256), 256, 0, stream>>>(H1, H1h,
                                                                    (long)E * H);

        // residual-before-skip
        launch_gemm<false, true, false, true, false, false, false, true>(
            stream, H1h, nullptr, base + 2 * HT, H,
            int_rb_b1 + (long)b * H, nullptr, nullptr, nullptr, nullptr, T1h, E);
        launch_gemm<false, true, false, true, false, true, true, true>(
            stream, T1h, nullptr, base + 3 * HT, H,
            int_rb_b2 + (long)b * H, nullptr, nullptr, H1, H2, H2h, E);
        // h = silu(h @ lin + b) + xe
        launch_gemm<false, true, false, true, false, true, true, true>(
            stream, H2h, nullptr, base + 4 * HT, H,
            int_lin_b + (long)b * H, nullptr, nullptr, X, H1, H1h, E);
        // residual-after-skip r=0
        launch_gemm<false, true, false, true, false, false, false, true>(
            stream, H1h, nullptr, base + 5 * HT, H,
            int_ra_b1 + (long)(b * 2 + 0) * H, nullptr, nullptr, nullptr,
            nullptr, T1h, E);
        launch_gemm<false, true, false, true, false, true, true, true>(
            stream, T1h, nullptr, base + 7 * HT, H,
            int_ra_b2 + (long)(b * 2 + 0) * H, nullptr, nullptr, H1, H2, H2h, E);
        // residual-after-skip r=1 -> new xe
        launch_gemm<false, true, false, true, false, false, false, true>(
            stream, H2h, nullptr, base + 6 * HT, H,
            int_ra_b1 + (long)(b * 2 + 1) * H, nullptr, nullptr, nullptr,
            nullptr, T1h, E);
        launch_gemm<false, true, false, true, false, true, true, true>(
            stream, T1h, nullptr, base + 8 * HT, H,
            int_ra_b2 + (long)(b * 2 + 1) * H, nullptr, nullptr, H2, X, Xh, E);

        out_block(b + 1);
    }
}
